// SpGAT_Conv_63917703299138
// MI455X (gfx1250) — matrix operand order repaired and checked by reading, not compiled, not run on hardware
//
#include <hip/hip_runtime.h>
#include <hip/hip_bf16.h>

typedef __bf16 bf16_t;
typedef __attribute__((ext_vector_type(4)))  __bf16 bf16x4;
typedef __attribute__((ext_vector_type(8)))  __bf16 bf16x8;
typedef __attribute__((ext_vector_type(16))) __bf16 bf16x16;
typedef __attribute__((ext_vector_type(4)))  float  fx4;
typedef __attribute__((ext_vector_type(8)))  float  fx8;

#define BM 128
#define BN 128
#define BK 32
#define LDSP 40                 // padded pitch (bf16): conflict-free b128 frag loads
#define ATILE (BM * LDSP)       // 5120 elems
#define BTILE (BN * LDSP)       // 5120 elems
#define BUFE  (ATILE + BTILE)   // elems per pipeline stage

enum { EPI_BF16T = 0, EPI_SCALE0 = 1, EPI_FINAL = 2 };

// C[M,N] = A[M,K](f32, row-major) x B[K,N], where B is supplied PRE-TRANSPOSED
// as BT[N,K] bf16 row-major. bf16 WMMA, f32 accumulate. Software-pipelined
// (global->VGPR prefetch across the WMMAs, ping-pong LDS, 1 barrier/K-step).
//
// EPI_BF16T : Cout = CT bf16 [N,M]  (transposed store: feeds the next GEMM's
//             BT operand; one b128 store per 16x16 tile per lane)
// EPI_SCALE0: Cout = f32 [M,N], a0 * acc          (a = softmax(alpha))
// EPI_FINAL : Cout = f32 [M,N], relu(max(lowIn, a1*acc) + bias[n])
template<int EPI>
__global__ __launch_bounds__(256)
void gemm_wmma_bf16(const float*  __restrict__ A,
                    const bf16_t* __restrict__ BT,
                    void*         __restrict__ Cout,
                    const float*  __restrict__ lowIn,
                    const float*  __restrict__ bias,
                    const float*  __restrict__ alpha,
                    int M, int N, int K)
{
    __shared__ bf16_t lds[2 * BUFE];   // 40 KB ping-pong

    const int t    = threadIdx.x;
    const int lane = t & 31;
    const int wave = t >> 5;
    const int wm   = wave & 3;      // 4 wave-rows  -> 32 M-rows each
    const int wn   = wave >> 2;     // 2 wave-cols  -> 64 N-cols each
    const int hi   = lane >> 4;     // half-wave select
    const int lm   = lane & 15;

    const int m_blk = blockIdx.x * BM;
    const int n_blk = blockIdx.y * BN;

    fx8 acc[2][4];
#pragma unroll
    for (int i = 0; i < 2; ++i)
#pragma unroll
        for (int j = 0; j < 4; ++j)
            acc[i][j] = (fx8){0.f,0.f,0.f,0.f,0.f,0.f,0.f,0.f};

    // ---- prefetch registers (one BK-stage of A and BT) ----
    fx4    ra[4];
    bf16x4 rb[4];

    auto loadG = [&](int k0) {
#pragma unroll
        for (int p = 0; p < 4; ++p) {
            int e   = p * 256 + t;          // unit id, 0..1023
            int row = e >> 3;               // 0..127
            int c4  = e & 7;                // 0..7
            ra[p] = *(const fx4*)(A + (size_t)(m_blk + row) * K + k0 + c4 * 4);
        }
#pragma unroll
        for (int p = 0; p < 4; ++p) {
            int e   = p * 256 + t;
            int row = e >> 3;               // n-row 0..127
            int c4  = e & 7;
            rb[p] = *(const bf16x4*)(BT + (size_t)(n_blk + row) * K + k0 + c4 * 4);
        }
    };

    auto storeL = [&](int buf) {
        bf16_t* la = &lds[buf * BUFE];
        bf16_t* lb = la + ATILE;
#pragma unroll
        for (int p = 0; p < 4; ++p) {
            int e   = p * 256 + t;
            int row = e >> 3;
            int c4  = e & 7;
            bf16x4 b = { (bf16_t)ra[p][0], (bf16_t)ra[p][1],
                         (bf16_t)ra[p][2], (bf16_t)ra[p][3] };
            *(bf16x4*)&la[row * LDSP + c4 * 4] = b;
        }
#pragma unroll
        for (int p = 0; p < 4; ++p) {
            int e   = p * 256 + t;
            int row = e >> 3;
            int c4  = e & 7;
            *(bf16x4*)&lb[row * LDSP + c4 * 4] = rb[p];
        }
    };

    // ---- pipeline prologue ----
    loadG(0);
    storeL(0);
    int cur = 0;

    for (int k0 = 0; k0 < K; k0 += BK) {
        const bool has_next = (k0 + BK) < K;
        if (has_next) loadG(k0 + BK);       // loads in flight across the WMMAs
        __syncthreads();                     // stage `cur` visible to all waves

        const bf16_t* la = &lds[cur * BUFE];
        const bf16_t* lb = la + ATILE;

        // A frags: lane<16 -> K {0-7,16-23}, lane>=16 -> K {8-15,24-31}
        bf16x16 afr[2];
#pragma unroll
        for (int mt = 0; mt < 2; ++mt) {
            int row = wm * 32 + mt * 16 + lm;
            bf16x8 lo = *(const bf16x8*)&la[row * LDSP + hi * 8];
            bf16x8 hg = *(const bf16x8*)&la[row * LDSP + 16 + hi * 8];
            afr[mt] = __builtin_shufflevector(lo, hg,
                0,1,2,3,4,5,6,7,8,9,10,11,12,13,14,15);
        }
        // B frags: lane<16 -> K {0-7,16-23}, lane>=16 -> K {8-15,24-31} at column lm
        bf16x16 bfr[4];
#pragma unroll
        for (int nt = 0; nt < 4; ++nt) {
            int row = wn * 64 + nt * 16 + lm;
            bf16x8 lo = *(const bf16x8*)&lb[row * LDSP + hi * 8];
            bf16x8 hg = *(const bf16x8*)&lb[row * LDSP + 16 + hi * 8];
            bfr[nt] = __builtin_shufflevector(lo, hg,
                0,1,2,3,4,5,6,7,8,9,10,11,12,13,14,15);
        }
#pragma unroll
        for (int mt = 0; mt < 2; ++mt)
#pragma unroll
            for (int nt = 0; nt < 4; ++nt)
                acc[mt][nt] = __builtin_amdgcn_wmma_f32_16x16x32_bf16(
                    false, afr[mt], false, bfr[nt],
                    (short)0, acc[mt][nt], false, false);

        if (has_next) storeL(cur ^ 1);       // waits loadcnt here, after WMMAs
        cur ^= 1;
    }

    // softmax(alpha)
    float a0 = 0.f, a1 = 0.f;
    if (EPI != EPI_BF16T) {
        float e0 = __expf(alpha[0]);
        float e1 = __expf(alpha[1]);
        float inv = 1.0f / (e0 + e1);
        a0 = e0 * inv;
        a1 = e1 * inv;
    }

    // ---- epilogue: lane lm = N col, VGPR r -> M row (r + 8*hi) ----
#pragma unroll
    for (int mt = 0; mt < 2; ++mt) {
#pragma unroll
        for (int nt = 0; nt < 4; ++nt) {
            int gm0 = m_blk + wm * 32 + mt * 16 + hi * 8;
            int gn  = n_blk + wn * 64 + nt * 16 + lm;
            if (EPI == EPI_BF16T) {
                // CT[gn][gm0 .. gm0+7]: one 16B store per lane per tile
                fx8 v = acc[mt][nt];
                bf16x8 o = { (bf16_t)v[0], (bf16_t)v[1], (bf16_t)v[2], (bf16_t)v[3],
                             (bf16_t)v[4], (bf16_t)v[5], (bf16_t)v[6], (bf16_t)v[7] };
                *(bf16x8*)((bf16_t*)Cout + (size_t)gn * M + gm0) = o;
            } else {
#pragma unroll
                for (int r = 0; r < 8; ++r) {
                    size_t idx = (size_t)(gm0 + r) * N + gn;
                    float v = acc[mt][nt][r];
                    if (EPI == EPI_SCALE0) {
                        ((float*)Cout)[idx] = a0 * v;
                    } else {
                        float o = fmaxf(lowIn[idx], a1 * v) + bias[gn];
                        ((float*)Cout)[idx] = fmaxf(o, 0.0f);
                    }
                }
            }
        }
    }
}

// WT[n][k] = bf16(W[k][n]) for 512x512 W (one-off, ~1 MB).
__global__ __launch_bounds__(256)
void transpose_w(const float* __restrict__ W, bf16_t* __restrict__ WT)
{
    __shared__ float tile[32][33];
    const int tx = threadIdx.x & 31;
    const int ty = threadIdx.x >> 5;        // 8 rows per pass
    const int bk = blockIdx.x * 32;         // k block
    const int bn = blockIdx.y * 32;         // n block
#pragma unroll
    for (int i = 0; i < 4; ++i) {
        int r = ty + i * 8;
        tile[r][tx] = W[(size_t)(bk + r) * 512 + bn + tx];
    }
    __syncthreads();
#pragma unroll
    for (int i = 0; i < 4; ++i) {
        int r = ty + i * 8;
        WT[(size_t)(bn + r) * 512 + bk + tx] = (bf16_t)tile[tx][r];
    }
}

extern "C" void kernel_launch(void* const* d_in, const int* in_sizes, int n_in,
                              void* d_out, int out_size, void* d_ws, size_t ws_size,
                              hipStream_t stream) {
    (void)in_sizes; (void)n_in; (void)out_size; (void)ws_size;
    const float* x  = (const float*)d_in[0];   // [8192,512]
    const float* w  = (const float*)d_in[1];   // [512,512]
    const float* al = (const float*)d_in[2];   // [2]
    const float* bi = (const float*)d_in[3];   // [512]
    const float* s0 = (const float*)d_in[4];   // [8192,2048]
    const float* s1 = (const float*)d_in[5];   // [2048,8192]
    const float* s2 = (const float*)d_in[6];   // [8192,6144]
    const float* s3 = (const float*)d_in[7];   // [6144,8192]

    const int N = 8192, KF = 2048, D = 512;
    const int NH = N - KF;                     // 6144

    // workspace layout (~34.1 MB): all intermediates transposed [D x *] bf16
    char* ws = (char*)d_ws;
    bf16_t* wT   = (bf16_t*)ws;                                     // [512,512]
    bf16_t* preT = (bf16_t*)(ws + (size_t)D * D * 2);               // [512,8192]
    bf16_t* t1T  = (bf16_t*)(ws + (size_t)D * (D + N) * 2);         // [512,2048]
    bf16_t* t3T  = (bf16_t*)(ws + (size_t)D * (D + N + KF) * 2);    // [512,6144]
    float*  low  = (float*) (ws + (size_t)D * (D + N + KF + NH) * 2); // [8192,512] f32

    dim3 blk(256);
    // 0) wT = transpose(bf16(W))
    transpose_w<<<dim3(D / 32, D / 32), blk, 0, stream>>>(w, wT);
    // 1) preT = (x @ W)^T           [512,8192]
    gemm_wmma_bf16<EPI_BF16T><<<dim3(N / BM, D / BN), blk, 0, stream>>>(
        x, wT, preT, nullptr, nullptr, nullptr, N, D, D);
    // 2) t1T = (s1 @ pre)^T         [512,2048]
    gemm_wmma_bf16<EPI_BF16T><<<dim3(KF / BM, D / BN), blk, 0, stream>>>(
        s1, preT, t1T, nullptr, nullptr, nullptr, KF, D, N);
    // 3) t3T = (s3 @ pre)^T         [512,6144]
    gemm_wmma_bf16<EPI_BF16T><<<dim3(NH / BM, D / BN), blk, 0, stream>>>(
        s3, preT, t3T, nullptr, nullptr, nullptr, NH, D, N);
    // 4) low = a0 * (s0 @ t1)       [8192,512] f32
    gemm_wmma_bf16<EPI_SCALE0><<<dim3(N / BM, D / BN), blk, 0, stream>>>(
        s0, t1T, low, nullptr, nullptr, al, N, D, KF);
    // 5) out = relu(max(low, a1 * (s2 @ t3)) + bias)
    gemm_wmma_bf16<EPI_FINAL><<<dim3(N / BM, D / BN), blk, 0, stream>>>(
        s2, t3T, (float*)d_out, low, bi, al, N, D, NH);
}
